// features_attention_73675868995609
// MI455X (gfx1250) — matrix operand order repaired and checked by reading, not compiled, not run on hardware
//
#include <hip/hip_runtime.h>

typedef __attribute__((ext_vector_type(16))) _Float16 v16h;
typedef __attribute__((ext_vector_type(8)))  _Float16 v8h;
typedef __attribute__((ext_vector_type(4)))  _Float16 v4h;
typedef __attribute__((ext_vector_type(8)))  float    v8f;

#define DEVINL __device__ __forceinline__

// Load a 16-element f16 fragment as two 8-element (16B) loads separated by
// `step` elements.  A-layout uses step=16 (K chunks 0-7 & 16-23 per half-wave),
// B-layout uses step=16 as well (same K order as A).
DEVINL v16h ld_frag(const _Float16* p, int step) {
  v8h lo = *(const v8h*)p;
  v8h hi = *(const v8h*)(p + step);
  return __builtin_shufflevector(lo, hi, 0, 1, 2, 3, 4, 5, 6, 7,
                                 8, 9, 10, 11, 12, 13, 14, 15);
}

DEVINL v8f wmma_f32_f16(v16h a, v16h b, v8f c) {
  // (neg_a, A, neg_b, B, c_mod, C, reuse_a, reuse_b)
  return __builtin_amdgcn_wmma_f32_16x16x32_f16(false, a, false, b, (short)0, c,
                                                false, false);
}

// CDNA5 async tensor path: global -> LDS DMA, tracked by ASYNCcnt.
// GVS addressing: mem_addr = SGPR64 base + per-lane VGPR byte offset.
// LDS destination = low 32 bits of the generic pointer (flat aperture rule).
DEVINL void async_ld_b128(unsigned lds_off, unsigned gofs, const void* base) {
  asm volatile("global_load_async_to_lds_b128 %0, %1, %2"
               :: "v"(lds_off), "v"(gofs), "s"(base) : "memory");
}
DEVINL void wait_asynccnt0() { asm volatile("s_wait_asynccnt 0x0" ::: "memory"); }

// ---------------------------------------------------------------- prep kernels
__global__ void cvt_f16(const float* __restrict__ x, _Float16* __restrict__ y,
                        size_t n) {
  size_t i = ((size_t)blockIdx.x * 256 + threadIdx.x) * 4;
  if (i < n) {
    float4 v = *(const float4*)(x + i);
    v4h o;
    o[0] = (_Float16)v.x; o[1] = (_Float16)v.y;
    o[2] = (_Float16)v.z; o[3] = (_Float16)v.w;
    *(v4h*)(y + i) = o;
  }
}

// W[K][N] f32 -> Wt[N][K] f16 (weights are tiny; coalescing irrelevant)
__global__ void wt_f16(const float* __restrict__ W, _Float16* __restrict__ Wt,
                       int K, int N) {
  int i = blockIdx.x * 256 + threadIdx.x;
  if (i < K * N) {
    int k = i / N;
    int n = i - k * N;
    Wt[(size_t)n * K + k] = (_Float16)W[i];
  }
}

// ---------------------------------------------------------------- GEMM kernel
// C[M,N] = act(A[M,K] @ Bt[N,K]^T + bias).  M,N multiples of 128, K of 32.
enum { ACT_NONE = 0, ACT_RELU = 1, ACT_SIG = 2 };
enum { ST_F16 = 0, ST_F16T = 1, ST_F32 = 2 };  // F16T: per-batch transposed (V^T)

template <int ACT, int ST>
__global__ __launch_bounds__(256) void gemm_kernel(
    const _Float16* __restrict__ A, const _Float16* __restrict__ Bt,
    const float* __restrict__ bias, void* __restrict__ Cv, int M, int N, int K) {
  constexpr int LDS_A = 40;  // 32 + 8 pad: conflict-free b128 fragment loads
  __shared__ __align__(16) _Float16 As[2][128 * LDS_A];
  __shared__ __align__(16) _Float16 Bs[2][128 * LDS_A];

  const int tid = threadIdx.x;
  const int lane = tid & 31;
  const int w = tid >> 5;         // 8 waves
  const int wm = w >> 1;          // 4 along M (32 rows each)
  const int wn = w & 1;           // 2 along N (64 cols each)
  const int m0 = blockIdx.y * 128;
  const int n0 = blockIdx.x * 128;
  const int r16 = lane & 15;
  const int hh = lane >> 4;

  // loader geometry: 512 8-elem chunks per tile / 256 threads = 2 each
  const int r0 = tid >> 2;            // chunk 0 row
  const int c0 = (tid & 3) * 8;       // chunk 0 col (elements)
  const int r1 = (tid + 256) >> 2;    // chunk 1 row
  const int c1 = c0;                  // (tid+256)&3 == tid&3

  v8f acc[2][4];
#pragma unroll
  for (int mi = 0; mi < 2; ++mi)
#pragma unroll
    for (int ni = 0; ni < 4; ++ni)
#pragma unroll
      for (int e = 0; e < 8; ++e) acc[mi][ni][e] = 0.0f;

  auto prefetch = [&](int buf, int k0) {
    async_ld_b128((unsigned)(uintptr_t)&As[buf][r0 * LDS_A + c0],
                  (unsigned)(((size_t)(m0 + r0) * K + k0 + c0) * 2), A);
    async_ld_b128((unsigned)(uintptr_t)&Bs[buf][r0 * LDS_A + c0],
                  (unsigned)(((size_t)(n0 + r0) * K + k0 + c0) * 2), Bt);
    async_ld_b128((unsigned)(uintptr_t)&As[buf][r1 * LDS_A + c1],
                  (unsigned)(((size_t)(m0 + r1) * K + k0 + c1) * 2), A);
    async_ld_b128((unsigned)(uintptr_t)&Bs[buf][r1 * LDS_A + c1],
                  (unsigned)(((size_t)(n0 + r1) * K + k0 + c1) * 2), Bt);
  };

  const int nk = K >> 5;
  prefetch(0, 0);
  int buf = 0;
  for (int t = 0; t < nk; ++t) {
    wait_asynccnt0();   // this wave's async tile writes have landed in LDS
    __syncthreads();    // ... and everyone else's; prior buf reads are done
    if (t + 1 < nk) prefetch(buf ^ 1, (t + 1) << 5);

    v16h af[2], bf[4];
#pragma unroll
    for (int mi = 0; mi < 2; ++mi)
      af[mi] = ld_frag(&As[buf][(wm * 32 + mi * 16 + r16) * LDS_A + hh * 8], 16);
#pragma unroll
    for (int ni = 0; ni < 4; ++ni)
      bf[ni] = ld_frag(&Bs[buf][(wn * 64 + ni * 16 + r16) * LDS_A + hh * 8], 16);
#pragma unroll
    for (int mi = 0; mi < 2; ++mi)
#pragma unroll
      for (int ni = 0; ni < 4; ++ni)
        acc[mi][ni] = wmma_f32_f16(af[mi], bf[ni], acc[mi][ni]);
    buf ^= 1;
  }

  // Epilogue: C layout => lane holds rows {vi + 8*hh}, col = r16 (coalesced).
#pragma unroll
  for (int mi = 0; mi < 2; ++mi) {
#pragma unroll
    for (int ni = 0; ni < 4; ++ni) {
      int col = n0 + wn * 64 + ni * 16 + r16;
      int rbase = m0 + wm * 32 + mi * 16 + hh * 8;
      float bv = bias[col];
#pragma unroll
      for (int vi = 0; vi < 8; ++vi) {
        float x = acc[mi][ni][vi] + bv;
        if (ACT == ACT_RELU) x = fmaxf(x, 0.0f);
        if (ACT == ACT_SIG) x = 1.0f / (1.0f + __expf(-x));
        int row = rbase + vi;
        if (ST == ST_F16) {
          ((_Float16*)Cv)[(size_t)row * N + col] = (_Float16)x;
        } else if (ST == ST_F32) {
          ((float*)Cv)[(size_t)row * N + col] = x;
        } else {  // ST_F16T: row = b*2048 + n2  ->  VT[b][col][n2]
          int bb = row >> 11, n2 = row & 2047;
          ((_Float16*)Cv)[((size_t)bb * 128 + col) * 2048 + n2] = (_Float16)x;
        }
      }
    }
  }
}

// ------------------------------------------------------------ flash attention
// attn = softmax(Q K^T) V   per batch.  Q,K row-major [2048][128] f16,
// V supplied transposed [128][2048] f16.  Out: [2048][128] f16.
__global__ __launch_bounds__(256) void flash_kernel(
    const _Float16* __restrict__ Qg, const _Float16* __restrict__ Kg,
    const _Float16* __restrict__ VTg, _Float16* __restrict__ Og) {
  constexpr int LDK = 136;  // 128 + 8 pad
  constexpr int LDV = 72;   // 64 + 8 pad
  constexpr int LDP = 72;
  __shared__ __align__(16) _Float16 Ks[2][64 * LDK];
  __shared__ __align__(16) _Float16 Vs[2][128 * LDV];
  __shared__ __align__(16) _Float16 Ps[8 * 16 * LDP];  // per-wave P slabs

  const int b = blockIdx.y;
  const int m0 = blockIdx.x * 128;
  const int tid = threadIdx.x;
  const int lane = tid & 31;
  const int w = tid >> 5;  // 8 waves x 16 q-rows
  const int r16 = lane & 15;
  const int hh = lane >> 4;

  const _Float16* q = Qg + (size_t)b * 2048 * 128;
  const _Float16* kg = Kg + (size_t)b * 2048 * 128;
  const _Float16* vt = VTg + (size_t)b * 128 * 2048;
  _Float16* outp = Og + (size_t)b * 2048 * 128;

  // Q fragments live in registers for the whole pass (A-layout, 4 K-steps)
  v16h qf[4];
  {
    const _Float16* qrow = q + (size_t)(m0 + w * 16 + r16) * 128;
#pragma unroll
    for (int ks = 0; ks < 4; ++ks) qf[ks] = ld_frag(qrow + ks * 32 + hh * 8, 16);
  }

  v8f o[8];
#pragma unroll
  for (int nf = 0; nf < 8; ++nf)
#pragma unroll
    for (int e = 0; e < 8; ++e) o[nf][e] = 0.0f;

  float rm[8], rl[8];
#pragma unroll
  for (int vi = 0; vi < 8; ++vi) { rm[vi] = -1e30f; rl[vi] = 0.0f; }

  _Float16* ps = &Ps[w * 16 * LDP];

  // async K/V block prefetch (4+4 b128 chunks per thread per block)
  auto prefetch = [&](int buf, int j0) {
#pragma unroll
    for (int i = 0; i < 4; ++i) {  // K block: 64 rows x 128 cols
      int c = tid + i * 256;
      int r = c >> 4;
      int c8 = (c & 15) * 8;
      async_ld_b128((unsigned)(uintptr_t)&Ks[buf][r * LDK + c8],
                    (unsigned)(((size_t)(j0 + r) * 128 + c8) * 2), kg);
    }
#pragma unroll
    for (int i = 0; i < 4; ++i) {  // V^T block: 128 rows x 64 cols
      int c = tid + i * 256;
      int r = c >> 3;
      int c8 = (c & 7) * 8;
      async_ld_b128((unsigned)(uintptr_t)&Vs[buf][r * LDV + c8],
                    (unsigned)(((size_t)r * 2048 + j0 + c8) * 2), vt);
    }
  };

  prefetch(0, 0);
  int buf = 0;
  for (int j0 = 0; j0 < 2048; j0 += 64) {
    wait_asynccnt0();
    __syncthreads();
    if (j0 + 64 < 2048) prefetch(buf ^ 1, j0 + 64);

    // S = Q K^T tile: 16 rows x 64 cols per wave
    v8f s[4];
#pragma unroll
    for (int ni = 0; ni < 4; ++ni)
#pragma unroll
      for (int e = 0; e < 8; ++e) s[ni][e] = 0.0f;
#pragma unroll
    for (int ks = 0; ks < 4; ++ks)
#pragma unroll
      for (int ni = 0; ni < 4; ++ni) {
        v16h bfr =
            ld_frag(&Ks[buf][(ni * 16 + r16) * LDK + ks * 32 + hh * 8], 16);
        s[ni] = wmma_f32_f16(qf[ks], bfr, s[ni]);
      }

    // online softmax (row stats: reduce across the 16 lanes of a half-wave)
    float nm[8];
#pragma unroll
    for (int vi = 0; vi < 8; ++vi) nm[vi] = rm[vi];
#pragma unroll
    for (int ni = 0; ni < 4; ++ni)
#pragma unroll
      for (int vi = 0; vi < 8; ++vi) nm[vi] = fmaxf(nm[vi], s[ni][vi]);
#pragma unroll
    for (int off = 1; off < 16; off <<= 1)
#pragma unroll
      for (int vi = 0; vi < 8; ++vi)
        nm[vi] = fmaxf(nm[vi], __shfl_xor(nm[vi], off, 32));
    float sc[8], rs[8];
#pragma unroll
    for (int vi = 0; vi < 8; ++vi) {
      sc[vi] = __expf(rm[vi] - nm[vi]);
      rm[vi] = nm[vi];
      rs[vi] = 0.0f;
    }
#pragma unroll
    for (int ni = 0; ni < 4; ++ni)
#pragma unroll
      for (int vi = 0; vi < 8; ++vi) {
        float e = __expf(s[ni][vi] - nm[vi]);
        s[ni][vi] = e;
        rs[vi] += e;
      }
#pragma unroll
    for (int off = 1; off < 16; off <<= 1)
#pragma unroll
      for (int vi = 0; vi < 8; ++vi) rs[vi] += __shfl_xor(rs[vi], off, 32);
#pragma unroll
    for (int vi = 0; vi < 8; ++vi) rl[vi] = rl[vi] * sc[vi] + rs[vi];
#pragma unroll
    for (int nf = 0; nf < 8; ++nf)
#pragma unroll
      for (int vi = 0; vi < 8; ++vi) o[nf][vi] *= sc[vi];

    // P: C-layout -> per-wave LDS slab -> reload in A-layout (wave-private,
    // ordered by DScnt within the wave; no barrier needed)
#pragma unroll
    for (int ni = 0; ni < 4; ++ni)
#pragma unroll
      for (int vi = 0; vi < 8; ++vi)
        ps[(vi + 8 * hh) * LDP + ni * 16 + r16] = (_Float16)s[ni][vi];

    // O += P V  (2 K-steps of 32 over the 64-wide j block)
#pragma unroll
    for (int js = 0; js < 2; ++js) {
      v16h pa = ld_frag(&ps[r16 * LDP + js * 32 + hh * 8], 16);
#pragma unroll
      for (int nf = 0; nf < 8; ++nf) {
        v16h vb =
            ld_frag(&Vs[buf][(nf * 16 + r16) * LDV + js * 32 + hh * 8], 16);
        o[nf] = wmma_f32_f16(pa, vb, o[nf]);
      }
    }
    buf ^= 1;
  }

  float inv[8];
#pragma unroll
  for (int vi = 0; vi < 8; ++vi) inv[vi] = 1.0f / rl[vi];
  const int rbase = m0 + w * 16 + 8 * hh;
#pragma unroll
  for (int nf = 0; nf < 8; ++nf)
#pragma unroll
    for (int vi = 0; vi < 8; ++vi)
      outp[(size_t)(rbase + vi) * 128 + nf * 16 + r16] =
          (_Float16)(o[nf][vi] * inv[vi]);
}

// -------------------------------------------------------------------- launch
extern "C" void kernel_launch(void* const* d_in, const int* in_sizes, int n_in,
                              void* d_out, int out_size, void* d_ws,
                              size_t ws_size, hipStream_t stream) {
  (void)in_sizes; (void)n_in; (void)out_size;
  const float* Xmain = (const float*)d_in[0];
  const float* Xfeat = (const float*)d_in[1];
  const float* We1 = (const float*)d_in[2];  const float* be1 = (const float*)d_in[3];
  const float* We2 = (const float*)d_in[4];  const float* be2 = (const float*)d_in[5];
  const float* Wq  = (const float*)d_in[6];  const float* bq  = (const float*)d_in[7];
  const float* Wk  = (const float*)d_in[8];  const float* bk  = (const float*)d_in[9];
  const float* Wv  = (const float*)d_in[10]; const float* bv  = (const float*)d_in[11];
  const float* Wd1 = (const float*)d_in[12]; const float* bd1 = (const float*)d_in[13];
  const float* Wd2 = (const float*)d_in[14]; const float* bd2 = (const float*)d_in[15];
  float* out = (float*)d_out;

  const int Bn = 16, N1 = 2048, H = 768, LAT = 128, MID = 256;
  const size_t M = (size_t)Bn * N1;  // 32768 rows (same for both features)

  size_t off = 0;
  auto alloc = [&](size_t bytes) -> _Float16* {
    _Float16* p = (_Float16*)((char*)d_ws + off);
    off += (bytes + 255) & ~(size_t)255;
    return p;
  };
  _Float16* XH  = alloc(M * H * 2);    // f16 input (reused: main, feature, attn)
  _Float16* HB  = alloc(M * MID * 2);  // hidden (reused)
  _Float16* MB  = alloc(M * LAT * 2);  // m latent
  _Float16* FB  = alloc(M * LAT * 2);  // f latent
  _Float16* QB  = alloc(M * LAT * 2);
  _Float16* KB  = alloc(M * LAT * 2);
  _Float16* VB  = alloc(M * LAT * 2);  // V^T [B][128][2048]
  _Float16* W1t = alloc((size_t)H * MID * 2);
  _Float16* W2t = alloc((size_t)MID * LAT * 2);
  _Float16* Wqt = alloc((size_t)LAT * LAT * 2);
  _Float16* Wkt = alloc((size_t)LAT * LAT * 2);
  _Float16* Wvt = alloc((size_t)LAT * LAT * 2);
  _Float16* Wd1t = alloc((size_t)LAT * MID * 2);
  _Float16* Wd2t = alloc((size_t)MID * H * 2);
  _Float16* AT = XH;  // attention output reuses XH (dead by then)
  if (ws_size < off) return;

  dim3 blk(256);
  // Weight transpose + f16 conversion
  wt_f16<<<dim3((H * MID + 255) / 256), blk, 0, stream>>>(We1, W1t, H, MID);
  wt_f16<<<dim3((MID * LAT + 255) / 256), blk, 0, stream>>>(We2, W2t, MID, LAT);
  wt_f16<<<dim3((LAT * LAT + 255) / 256), blk, 0, stream>>>(Wq, Wqt, LAT, LAT);
  wt_f16<<<dim3((LAT * LAT + 255) / 256), blk, 0, stream>>>(Wk, Wkt, LAT, LAT);
  wt_f16<<<dim3((LAT * LAT + 255) / 256), blk, 0, stream>>>(Wv, Wvt, LAT, LAT);
  wt_f16<<<dim3((LAT * MID + 255) / 256), blk, 0, stream>>>(Wd1, Wd1t, LAT, MID);
  wt_f16<<<dim3((MID * H + 255) / 256), blk, 0, stream>>>(Wd2, Wd2t, MID, H);

  const dim3 gEnc1(MID / 128, M / 128), gEnc2(LAT / 128, M / 128);
  const dim3 gQKV(LAT / 128, M / 128);
  const dim3 gDec1(MID / 128, M / 128), gDec2(H / 128, M / 128);
  const size_t nX = M * H;

  // main encode
  cvt_f16<<<dim3((unsigned)(nX / 4 / 256)), blk, 0, stream>>>(Xmain, XH, nX);
  gemm_kernel<ACT_RELU, ST_F16><<<gEnc1, blk, 0, stream>>>(XH, W1t, be1, HB, (int)M, MID, H);
  gemm_kernel<ACT_RELU, ST_F16><<<gEnc2, blk, 0, stream>>>(HB, W2t, be2, MB, (int)M, LAT, MID);
  // feature encode
  cvt_f16<<<dim3((unsigned)(nX / 4 / 256)), blk, 0, stream>>>(Xfeat, XH, nX);
  gemm_kernel<ACT_RELU, ST_F16><<<gEnc1, blk, 0, stream>>>(XH, W1t, be1, HB, (int)M, MID, H);
  gemm_kernel<ACT_RELU, ST_F16><<<gEnc2, blk, 0, stream>>>(HB, W2t, be2, FB, (int)M, LAT, MID);
  // q, k, v^T
  gemm_kernel<ACT_NONE, ST_F16><<<gQKV, blk, 0, stream>>>(MB, Wqt, bq, QB, (int)M, LAT, LAT);
  gemm_kernel<ACT_NONE, ST_F16><<<gQKV, blk, 0, stream>>>(FB, Wkt, bk, KB, (int)M, LAT, LAT);
  gemm_kernel<ACT_NONE, ST_F16T><<<gQKV, blk, 0, stream>>>(FB, Wvt, bv, VB, (int)M, LAT, LAT);
  // fused attention
  flash_kernel<<<dim3(N1 / 128, Bn), blk, 0, stream>>>(QB, KB, VB, AT);
  // decode
  gemm_kernel<ACT_RELU, ST_F16><<<gDec1, blk, 0, stream>>>(AT, Wd1t, bd1, HB, (int)M, MID, LAT);
  gemm_kernel<ACT_SIG, ST_F32><<<gDec2, blk, 0, stream>>>(HB, Wd2t, bd2, out, (int)M, H, MID);
}
